// DeltaNet_31877247271514
// MI455X (gfx1250) — matrix operand order repaired and checked by reading, not compiled, not run on hardware
//
#include <hip/hip_runtime.h>
#include <hip/hip_bf16.h>
#include <math.h>

typedef _Float16 half8 __attribute__((ext_vector_type(8)));
typedef _Float16 v16h  __attribute__((ext_vector_type(16)));
typedef float    v8f   __attribute__((ext_vector_type(8)));

constexpr int Bb = 2, Ll = 1024, HID = 1024, NH = 4, DK = 256, DV = 256;
constexpr int CHUNK = 32, NCHUNK = Ll / CHUNK;   // 32
constexpr int BL = Bb * Ll;                       // 2048
constexpr int ROWS = BL * NH;                     // 8192
constexpr int GH = 512;

#define SHUF16(lo, hi) __builtin_shufflevector(lo, hi, 0,1,2,3,4,5,6,7,8,9,10,11,12,13,14,15)

static __device__ __forceinline__ v8f zero8() {
  v8f z;
#pragma unroll
  for (int i = 0; i < 8; ++i) z[i] = 0.f;
  return z;
}
static __device__ __forceinline__ v8f wmma16(v16h a, v16h b, v8f c) {
  // D = A(16x32 f16) x B(32x16 f16) + C(16x16 f32)
  return __builtin_amdgcn_wmma_f32_16x16x32_f16(false, a, false, b, (short)0, c, false, false);
}

// A operand (16x32 f16) from row-major [M, lda] f16.
// Lane l: row m = m0+(l&15); halfs: K = k0 + 16*(l>>4) + i
static __device__ __forceinline__ v16h load_a_f16(const _Float16* A, int lda, int m0, int k0, int lane) {
  int m = m0 + (lane & 15);
  int kb = k0 + ((lane >> 4) << 4);
  const _Float16* p = A + (size_t)m * lda + kb;
  half8 lo = *(const half8*)p;
  half8 hi = *(const half8*)(p + 8);
  return SHUF16(lo, hi);
}
// A operand where logical A = K^T (A[m][k] = Krm[k][m]); Krm row-major [K, ld] f16.
static __device__ __forceinline__ v16h load_aT_f16(const _Float16* Krm, int ld, int m0, int k0, int lane) {
  int m = m0 + (lane & 15);
  int kb = k0 + ((lane >> 4) << 4);
  v16h r;
#pragma unroll
  for (int i = 0; i < 8; ++i)  r[i] = Krm[(size_t)(kb + i) * ld + m];
#pragma unroll
  for (int i = 8; i < 16; ++i) r[i] = Krm[(size_t)(kb + i) * ld + m];
  return r;
}
// B operand (32x16 f16) from K-contiguous storage BT[n][k], row-major [N, ldb] f16.
// Lane l: col n = n0+(l&15); halfs: K = k0 + 16*(l>>4) + i
static __device__ __forceinline__ v16h load_b_f16(const _Float16* BT, int ldb, int n0, int k0, int lane) {
  int n = n0 + (lane & 15);
  const _Float16* p = BT + (size_t)n * ldb + k0 + ((lane >> 4) << 4);
  half8 lo = *(const half8*)p;
  half8 hi = *(const half8*)(p + 8);
  return SHUF16(lo, hi);
}
// B operand from row-major [K, ld] f16 (strided column gather).
static __device__ __forceinline__ v16h load_b_rm_f16(const _Float16* Brm, int ld, int k0, int n0, int lane) {
  int n = n0 + (lane & 15);
  int kb = k0 + ((lane >> 4) << 4);
  v16h r;
#pragma unroll
  for (int i = 0; i < 16; ++i) r[i] = Brm[(size_t)(kb + i) * ld + n];
  return r;
}
// B operand from row-major [K, ld] f32 (e.g. state S in LDS), converting to f16.
static __device__ __forceinline__ v16h load_b_rm_f32(const float* Brm, int ld, int k0, int n0, int lane) {
  int n = n0 + (lane & 15);
  int kb = k0 + ((lane >> 4) << 4);
  v16h r;
#pragma unroll
  for (int i = 0; i < 16; ++i) r[i] = (_Float16)Brm[(size_t)(kb + i) * ld + n];
  return r;
}

static __device__ __forceinline__ float wave_sum(float v) {
#pragma unroll
  for (int o = 16; o > 0; o >>= 1) v += __shfl_xor(v, o, 32);
  return v;
}
static __device__ __forceinline__ float wave_max(float v) {
#pragma unroll
  for (int o = 16; o > 0; o >>= 1) v = fmaxf(v, __shfl_xor(v, o, 32));
  return v;
}

// ---------------------------------------------------------------- conversions
__global__ void k_cvt_f16(const float* __restrict__ src, _Float16* __restrict__ dst, int n) {
  int i = blockIdx.x * blockDim.x + threadIdx.x;
  if (i < n) dst[i] = (_Float16)src[i];
}
// src [rows][cols] f32  ->  dst [cols][rows] f16 (K-contiguous weight for B operand)
__global__ void k_cvt_f16_T(const float* __restrict__ src, _Float16* __restrict__ dst, int rows, int cols) {
  int i = blockIdx.x * blockDim.x + threadIdx.x;
  if (i >= rows * cols) return;
  int r = i / cols, c = i - r * cols;
  dst[(size_t)c * rows + r] = (_Float16)src[i];
}

// ---------------------------------------------------------------- generic WMMA GEMM
// C[M,N] (f32) = (accflag ? C : 0) + A[M,K](f16, row r of C reads A row r/rowdiv) @ B
// B given K-contiguous: BT[N][ldb], logical B[k][n] = BT[n][koff + k].
// Each wave computes a 16x64 tile (A reuse x4), K-loop software-pipelined so the
// next k-step's global_load_b128 clauses issue before the current WMMA group.
__global__ void k_gemm(const _Float16* __restrict__ A, int lda, int rowdiv,
                       const _Float16* __restrict__ BT, int ldb, int koff,
                       float* __restrict__ C, int ldc, int M, int N, int K, int accflag) {
  int lane = threadIdx.x & 31, wave = threadIdx.x >> 5;
  int tiles_n = N >> 6;
  int tile = blockIdx.x * (blockDim.x >> 5) + wave;
  int total = (M >> 4) * tiles_n;
  if (tile >= total) return;
  int tm = tile / tiles_n, tn = tile - tm * tiles_n;
  int mlane = tm * 16 + (lane & 15);
  const _Float16* Arow = A + (size_t)(mlane / rowdiv) * lda + ((lane >> 4) << 4);
  const _Float16* Brow[4];
#pragma unroll
  for (int nt = 0; nt < 4; ++nt)
    Brow[nt] = BT + (size_t)(tn * 64 + nt * 16 + (lane & 15)) * ldb + koff + ((lane >> 4) << 4);

  auto ldA = [&](int kk) -> v16h {
    half8 lo = *(const half8*)(Arow + kk);
    half8 hi = *(const half8*)(Arow + kk + 8);
    return SHUF16(lo, hi);
  };
  auto ldB = [&](int nt, int kk) -> v16h {
    half8 lo = *(const half8*)(Brow[nt] + kk);
    half8 hi = *(const half8*)(Brow[nt] + kk + 8);
    return SHUF16(lo, hi);
  };

  int mb = tm * 16 + ((lane >> 4) << 3);
  int nb = tn * 64 + (lane & 15);
  v8f acc[4];
#pragma unroll
  for (int nt = 0; nt < 4; ++nt) acc[nt] = zero8();
  if (accflag) {
#pragma unroll
    for (int nt = 0; nt < 4; ++nt)
#pragma unroll
      for (int r = 0; r < 8; ++r)
        acc[nt][r] = C[(size_t)(mb + r) * ldc + nb + nt * 16];
  }

  // prologue: k-step 0 in registers
  v16h av = ldA(0);
  v16h bv0 = ldB(0, 0), bv1 = ldB(1, 0), bv2 = ldB(2, 0), bv3 = ldB(3, 0);
  for (int kk = 32; kk < K; kk += 32) {
    // issue next k-step's loads before consuming the current one
    v16h av_n = ldA(kk);
    v16h bn0 = ldB(0, kk), bn1 = ldB(1, kk), bn2 = ldB(2, kk), bn3 = ldB(3, kk);
    __builtin_prefetch(Arow + kk + 128, 0, 3);
    acc[0] = wmma16(av, bv0, acc[0]);
    acc[1] = wmma16(av, bv1, acc[1]);
    acc[2] = wmma16(av, bv2, acc[2]);
    acc[3] = wmma16(av, bv3, acc[3]);
    av = av_n;
    bv0 = bn0; bv1 = bn1; bv2 = bn2; bv3 = bn3;
  }
  acc[0] = wmma16(av, bv0, acc[0]);
  acc[1] = wmma16(av, bv1, acc[1]);
  acc[2] = wmma16(av, bv2, acc[2]);
  acc[3] = wmma16(av, bv3, acc[3]);

#pragma unroll
  for (int nt = 0; nt < 4; ++nt)
#pragma unroll
    for (int r = 0; r < 8; ++r)
      C[(size_t)(mb + r) * ldc + nb + nt * 16] = acc[nt][r];
}

// ---------------------------------------------------------------- beta = sigmoid(x@Wb)
__global__ void k_beta(const float* __restrict__ x, const float* __restrict__ Wb,
                       float* __restrict__ beta) {
  int bl = blockIdx.x;                 // block 128 = 4 waves, wave h
  int lane = threadIdx.x & 31, h = threadIdx.x >> 5;
  const float* xr = x + (size_t)bl * HID;
  float s = 0.f;
  for (int j = lane; j < HID; j += 32) s += xr[j] * Wb[j * NH + h];
  s = wave_sum(s);
  if (lane == 0) beta[(size_t)bl * NH + h] = 1.f / (1.f + __expf(-s));
}

// ---------------------------------------------------------------- causal depthwise conv (k=4) + silu
__global__ void k_conv_silu(const float* __restrict__ xin, const float* __restrict__ w,
                            float* __restrict__ y, int C) {
  int i = blockIdx.x * blockDim.x + threadIdx.x;    // over B*L*C
  if (i >= BL * C) return;
  int c = i % C;
  int bl = i / C;
  int l = bl & (Ll - 1);
  float acc = 0.f;
#pragma unroll
  for (int t = 0; t < 4; ++t) {
    int ls = l - 3 + t;
    if (ls >= 0) acc += w[c * 4 + t] * xin[(size_t)(bl - 3 + t) * C + c];
  }
  y[i] = acc / (1.f + __expf(-acc));
}

// ---------------------------------------------------------------- per-head depthwise causal FIR
__global__ void k_fir(const float* __restrict__ v, const float* __restrict__ f,
                      float* __restrict__ y, int KF) {
  int i = blockIdx.x * blockDim.x + threadIdx.x;    // over B*L*NH*DV
  if (i >= ROWS * DV) return;
  int d = i % DV;
  int h = (i / DV) % NH;
  int bl = i / (NH * DV);
  int l = bl & (Ll - 1);
  const float* fp = f + ((size_t)h * DV + d) * KF;
  float acc = 0.f;
  for (int t = 0; t < KF; ++t) {
    int ls = l - (KF - 1) + t;
    if (ls >= 0) acc += fp[t] * v[(size_t)i + (long)(ls - l) * (NH * DV)];
  }
  y[i] = acc;
}

// ---------------------------------------------------------------- delta-rule per-chunk precompute
// One block per (b,h,chunk): l2norm q/k, scale by beta, A = -tril(kb@kn^T,-1) via WMMA,
// forward substitution for u = T@v, w = T@kb; emit f16 q,k,u,w tiles.
__global__ void k_delta_pre(const float* __restrict__ qc, const float* __restrict__ kc,
                            const float* __restrict__ vc, const float* __restrict__ beta,
                            _Float16* __restrict__ q16, _Float16* __restrict__ k16,
                            _Float16* __restrict__ u16g, _Float16* __restrict__ w16g) {
  extern __shared__ char smem[];
  _Float16* q16s = (_Float16*)smem;                 // 32*256 f16
  _Float16* k16s = q16s + 32 * 256;
  _Float16* kb16 = k16s + 32 * 256;
  float* uv = (float*)(kb16 + 32 * 256);            // 32*256 f32 (v*beta -> u)
  float* wk = uv + 32 * 256;                        // 32*256 f32 (kb   -> w)
  float* Am = wk + 32 * 256;                        // 32*32 f32 (strictly-lower, negated)
  int tid = threadIdx.x, lane = tid & 31, wave = tid >> 5;
  int blk = blockIdx.x;
  int chunk = blk & (NCHUNK - 1);
  int bh = blk >> 5;
  int b = bh >> 2, h = bh & 3;
  size_t gbase = ((size_t)bh * NCHUNK + chunk) * (32 * 256);

  for (int rr = 0; rr < 4; ++rr) {
    int r = wave * 4 + rr;
    int l = chunk * 32 + r;
    size_t rowoff = ((size_t)(b * Ll + l)) * (NH * DK) + (size_t)h * DK;
    const float* qrow = qc + rowoff;
    const float* krow = kc + rowoff;
    const float* vrow = vc + rowoff;
    float bta = beta[(size_t)(b * Ll + l) * NH + h];
    float qv[8], kv[8], vv[8];
    float qss = 0.f, kss = 0.f;
#pragma unroll
    for (int j = 0; j < 8; ++j) {
      int d = lane + 32 * j;
      qv[j] = qrow[d]; kv[j] = krow[d]; vv[j] = vrow[d];
      qss += qv[j] * qv[j]; kss += kv[j] * kv[j];
    }
    qss = wave_sum(qss); kss = wave_sum(kss);
    float rq = rsqrtf(qss + 1e-6f), rk = rsqrtf(kss + 1e-6f);
#pragma unroll
    for (int j = 0; j < 8; ++j) {
      int d = lane + 32 * j;
      float qn = qv[j] * rq, kn = kv[j] * rk;
      q16s[r * 256 + d] = (_Float16)qn;
      k16s[r * 256 + d] = (_Float16)kn;
      q16[gbase + r * 256 + d] = (_Float16)qn;
      k16[gbase + r * 256 + d] = (_Float16)kn;
      float kbv = kn * bta;
      kb16[r * 256 + d] = (_Float16)kbv;
      wk[r * 256 + d] = kbv;
      uv[r * 256 + d] = vv[j] * bta;
    }
  }
  __syncthreads();
  if (wave < 4) {                                   // A = kb @ kn^T  (32x32, K=256)
    int tm = wave >> 1, tn = wave & 1;
    v8f aa = zero8();
    for (int kt = 0; kt < 8; ++kt) {
      v16h aKB = load_a_f16(kb16, 256, tm * 16, kt * 32, lane);
      v16h bK  = load_b_f16(k16s, 256, tn * 16, kt * 32, lane);  // B[k][n] = kn[n][k]
      aa = wmma16(aKB, bK, aa);
    }
    int n = tn * 16 + (lane & 15);
#pragma unroll
    for (int r = 0; r < 8; ++r) {
      int m = tm * 16 + r + ((lane >> 4) << 3);
      Am[m * 32 + n] = (n < m) ? -aa[r] : 0.f;      // A = -tril(.,-1)
    }
  }
  __syncthreads();
  {                                                 // (I-A)u = v ; (I-A)w = kb  (column-per-thread)
    int t = tid;                                    // column d, 256 threads
    for (int i = 1; i < 32; ++i) {
      float au = uv[i * 256 + t], aw = wk[i * 256 + t];
      for (int j = 0; j < i; ++j) {
        float a = Am[i * 32 + j];
        au += a * uv[j * 256 + t];
        aw += a * wk[j * 256 + t];
      }
      uv[i * 256 + t] = au;
      wk[i * 256 + t] = aw;
    }
    for (int i = 0; i < 32; ++i) {
      u16g[gbase + i * 256 + t] = (_Float16)uv[i * 256 + t];
      w16g[gbase + i * 256 + t] = (_Float16)wk[i * 256 + t];
    }
  }
}

// ---------------------------------------------------------------- delta-rule serial scan
// One block per (b,h); S[256][256] f32 resident in LDS across the 32-chunk scan.
__global__ void k_delta_scan(const _Float16* __restrict__ q16, const _Float16* __restrict__ k16,
                             const _Float16* __restrict__ u16g, const _Float16* __restrict__ w16g,
                             float* __restrict__ dlt) {
  extern __shared__ char smem[];
  float* S = (float*)smem;                               // 256*256 f32 = 256KB
  _Float16* uls = (_Float16*)(smem + 256 * 256 * 4);     // 32*256 f16
  _Float16* att = (_Float16*)(smem + 256 * 256 * 4 + 32 * 256 * 2); // 32*32 f16
  int tid = threadIdx.x, lane = tid & 31, wave = tid >> 5;
  int bh = blockIdx.x;
  int b = bh >> 2, h = bh & 3;
  const _Float16* qb = q16 + (size_t)bh * (Ll * 256);
  const _Float16* kb = k16 + (size_t)bh * (Ll * 256);
  const _Float16* ub = u16g + (size_t)bh * (Ll * 256);
  const _Float16* wb = w16g + (size_t)bh * (Ll * 256);
  float* ob = dlt + ((size_t)b * Ll * NH + h) * DV;

  for (int i = tid; i < 256 * 256; i += 256) S[i] = 0.f;
  __syncthreads();

  for (int c = 0; c < NCHUNK; ++c) {
    const _Float16* qcp = qb + (size_t)c * (32 * 256);
    const _Float16* kcp = kb + (size_t)c * (32 * 256);
    const _Float16* ucp = ub + (size_t)c * (32 * 256);
    const _Float16* wcp = wb + (size_t)c * (32 * 256);
    v8f accq[2][2];
    // Phase A: qS = q@S, wS = w@S  (wave owns 32 columns); Phase B: u = u_pre - wS -> LDS f16
    for (int ntl = 0; ntl < 2; ++ntl) {
      int n0 = wave * 32 + ntl * 16;
      v8f aq0 = zero8(), aq1 = zero8(), aw0 = zero8(), aw1 = zero8();
      for (int kt = 0; kt < 8; ++kt) {
        v16h bS = load_b_rm_f32(S, 256, kt * 32, n0, lane);
        v16h a;
        a = load_a_f16(qcp, 256, 0,  kt * 32, lane); aq0 = wmma16(a, bS, aq0);
        a = load_a_f16(qcp, 256, 16, kt * 32, lane); aq1 = wmma16(a, bS, aq1);
        a = load_a_f16(wcp, 256, 0,  kt * 32, lane); aw0 = wmma16(a, bS, aw0);
        a = load_a_f16(wcp, 256, 16, kt * 32, lane); aw1 = wmma16(a, bS, aw1);
      }
      accq[ntl][0] = aq0;
      accq[ntl][1] = aq1;
      int nn = n0 + (lane & 15);
#pragma unroll
      for (int r = 0; r < 8; ++r) {
        int mr = r + ((lane >> 4) << 3);
        uls[(size_t)mr * 256 + nn]        = (_Float16)((float)ucp[(size_t)mr * 256 + nn] - aw0[r]);
        uls[(size_t)(16 + mr) * 256 + nn] = (_Float16)((float)ucp[(size_t)(16 + mr) * 256 + nn] - aw1[r]);
      }
    }
    // attn = tril(q @ k^T) incl. diagonal (waves 0..3)
    if (wave < 4) {
      int tm = wave >> 1, tn = wave & 1;
      v8f aa = zero8();
      for (int kt = 0; kt < 8; ++kt) {
        v16h aQ = load_a_f16(qcp, 256, tm * 16, kt * 32, lane);
        v16h bK = load_b_f16(kcp, 256, tn * 16, kt * 32, lane);   // B[k][n] = k[n][k]
        aa = wmma16(aQ, bK, aa);
      }
      int n = tn * 16 + (lane & 15);
#pragma unroll
      for (int r = 0; r < 8; ++r) {
        int m = tm * 16 + r + ((lane >> 4) << 3);
        att[m * 32 + n] = (n <= m) ? (_Float16)aa[r] : (_Float16)0.f;
      }
    }
    __syncthreads();
    // Phase C: o = qS + attn @ u  (K=32, single WMMA depth)
    for (int ntl = 0; ntl < 2; ++ntl) {
      int n0 = wave * 32 + ntl * 16;
      v16h bU = load_b_rm_f16(uls, 256, 0, n0, lane);
#pragma unroll
      for (int mt = 0; mt < 2; ++mt) {
        v16h aA = load_a_f16(att, 32, mt * 16, 0, lane);
        v8f o = wmma16(aA, bU, accq[ntl][mt]);
        int n = n0 + (lane & 15);
#pragma unroll
        for (int r = 0; r < 8; ++r) {
          int row = c * 32 + mt * 16 + r + ((lane >> 4) << 3);
          ob[(size_t)row * (NH * DV) + n] = o[r];
        }
      }
    }
    // Phase D: S += k^T @ u  (wave owns 32 rows of S; C operand streamed from LDS S)
    {
      int m0 = wave * 32;
      v16h at0 = load_aT_f16(kcp, 256, m0,      0, lane);
      v16h at1 = load_aT_f16(kcp, 256, m0 + 16, 0, lane);
      for (int nt = 0; nt < 16; ++nt) {
        v16h bU = load_b_rm_f16(uls, 256, 0, nt * 16, lane);
#pragma unroll
        for (int mt = 0; mt < 2; ++mt) {
          int mbase = m0 + mt * 16 + ((lane >> 4) << 3);
          int n = nt * 16 + (lane & 15);
          v8f cS;
#pragma unroll
          for (int r = 0; r < 8; ++r) cS[r] = S[(size_t)(mbase + r) * 256 + n];
          v8f dS = wmma16(mt ? at1 : at0, bU, cS);
#pragma unroll
          for (int r = 0; r < 8; ++r) S[(size_t)(mbase + r) * 256 + n] = dS[r];
        }
      }
    }
    __syncthreads();
  }
}

// ---------------------------------------------------------------- gate: column sums of stat blocks of gW1
__global__ void k_colsum(const float* __restrict__ gW1, float* __restrict__ uvec) {
  int t = blockIdx.x * blockDim.x + threadIdx.x;    // 12*512
  if (t >= 12 * GH) return;
  int vi = t >> 9, j = t & (GH - 1);
  int br = vi / 3, s = vi % 3;
  int row0 = HID + 768 * br + 256 * s;
  float acc = 0.f;
  for (int r = 0; r < 256; ++r) acc += gW1[(size_t)(row0 + r) * GH + j];
  uvec[t] = acc;
}

// ---------------------------------------------------------------- branch stats (mean / rms / max)
__global__ void k_stats(const float* __restrict__ b0, const float* __restrict__ b1,
                        const float* __restrict__ b2, const float* __restrict__ b3,
                        float* __restrict__ stats) {
  int bl = blockIdx.x;
  int lane = threadIdx.x & 31, wave = threadIdx.x >> 5;
  const float* bp[4] = {b0, b1, b2, b3};
  for (int ii = 0; ii < 2; ++ii) {
    int combo = wave * 2 + ii;                      // 16 combos = NH x 4 branches
    int h = combo >> 2, br = combo & 3;
    const float* p = bp[br] + ((size_t)bl * NH + h) * DV;
    float s = 0.f, ss = 0.f, mx = -3.4e38f;
    for (int j = lane; j < DV; j += 32) {
      float v = p[j];
      s += v; ss += v * v; mx = fmaxf(mx, v);
    }
    s = wave_sum(s); ss = wave_sum(ss); mx = wave_max(mx);
    if (lane == 0) {
      float* o = stats + ((size_t)bl * NH + h) * 12 + br * 3;
      o[0] = s * (1.f / DV);
      o[1] = sqrtf(fmaxf(ss * (1.f / DV), 1e-8f));
      o[2] = mx;
    }
  }
}

// ---------------------------------------------------------------- hmid init: bias + rank-1 stat terms
__global__ void k_hmid_init(const float* __restrict__ stats, const float* __restrict__ uvec,
                            const float* __restrict__ gb1, float* __restrict__ hmid) {
  int i = blockIdx.x * blockDim.x + threadIdx.x;    // ROWS*512
  if (i >= ROWS * GH) return;
  int row = i >> 9, j = i & (GH - 1);
  const float* st = stats + (size_t)row * 12;
  float acc = gb1[j];
#pragma unroll
  for (int br = 0; br < 4; ++br) {
    acc += st[br * 3 + 0] * uvec[(br * 3 + 0) * GH + j];
    acc += st[br * 3 + 1] * uvec[(br * 3 + 1) * GH + j];
    acc += st[br * 3 + 2] * uvec[(br * 3 + 2) * GH + j];
  }
  hmid[i] = acc;
}

// ---------------------------------------------------------------- gate head: gelu, logits, softmax, mix, rmsnorm
__global__ void k_gate_mix(const float* __restrict__ hmid, const float* __restrict__ gW2,
                           const float* __restrict__ gb2, const float* __restrict__ temp,
                           const float* __restrict__ epsf, const float* __restrict__ onw,
                           const float* __restrict__ fs, const float* __restrict__ fl,
                           const float* __restrict__ dl, const float* __restrict__ vv,
                           _Float16* __restrict__ omix) {
  int bl = blockIdx.x;                              // block 128 = 4 waves, wave h
  int lane = threadIdx.x & 31, h = threadIdx.x >> 5;
  size_t row = (size_t)bl * NH + h;
  const float* hm = hmid + row * GH;
  float l0 = 0.f, l1 = 0.f, l2 = 0.f, l3 = 0.f;
  for (int j = lane; j < GH; j += 32) {
    float x = hm[j];
    float g = 0.5f * x * (1.f + erff(x * 0.70710678118f));
    const float* w2 = gW2 + j * 4;
    l0 += g * w2[0]; l1 += g * w2[1]; l2 += g * w2[2]; l3 += g * w2[3];
  }
  l0 = wave_sum(l0) + gb2[0];
  l1 = wave_sum(l1) + gb2[1];
  l2 = wave_sum(l2) + gb2[2];
  l3 = wave_sum(l3) + gb2[3];
  float t = fminf(fmaxf(temp[h], 0.2f), 10.f);
  l0 /= t; l1 /= t; l2 /= t; l3 /= t;
  float m = fmaxf(fmaxf(l0, l1), fmaxf(l2, l3));
  float e0 = __expf(l0 - m), e1 = __expf(l1 - m), e2 = __expf(l2 - m), e3 = __expf(l3 - m);
  float es = e0 + e1 + e2 + e3;
  float w0 = e0 / es, w1 = e1 / es, w2_ = e2 / es, w3 = e3 / es;
  w0 = fmaxf(w0, fminf(fmaxf(epsf[h * 4 + 0], 1e-7f), 0.1f));
  w1 = fmaxf(w1, fminf(fmaxf(epsf[h * 4 + 1], 1e-7f), 0.1f));
  w2_ = fmaxf(w2_, fminf(fmaxf(epsf[h * 4 + 2], 1e-7f), 0.1f));
  w3 = fmaxf(w3, fminf(fmaxf(epsf[h * 4 + 3], 1e-7f), 0.1f));
  float wsum = w0 + w1 + w2_ + w3;
  w0 /= wsum; w1 /= wsum; w2_ /= wsum; w3 /= wsum;
  const float* fsp = fs + row * DV;
  const float* flp = fl + row * DV;
  const float* dlp = dl + row * DV;
  const float* vp_ = vv + row * DV;
  float ov[8], ssq = 0.f;
#pragma unroll
  for (int jj = 0; jj < 8; ++jj) {
    int d = lane + 32 * jj;
    float o = w0 * fsp[d] + w1 * flp[d] + w2_ * dlp[d] + w3 * vp_[d];
    ov[jj] = o; ssq += o * o;
  }
  ssq = wave_sum(ssq);
  float sc = rsqrtf(ssq * (1.f / DV) + 1e-5f);
#pragma unroll
  for (int jj = 0; jj < 8; ++jj) {
    int d = lane + 32 * jj;
    omix[(size_t)bl * (NH * DV) + h * DV + d] = (_Float16)(ov[jj] * sc * onw[d]);
  }
}

// ---------------------------------------------------------------- host launcher
extern "C" void kernel_launch(void* const* d_in, const int* in_sizes, int n_in,
                              void* d_out, int out_size, void* d_ws, size_t ws_size,
                              hipStream_t stream) {
  (void)in_sizes; (void)n_in; (void)out_size; (void)ws_size;
  const float* x     = (const float*)d_in[0];
  const float* Wq    = (const float*)d_in[1];
  const float* Wk    = (const float*)d_in[2];
  const float* Wv    = (const float*)d_in[3];
  const float* Wb    = (const float*)d_in[4];
  const float* qc_w  = (const float*)d_in[5];
  const float* kc_w  = (const float*)d_in[6];
  const float* vc_w  = (const float*)d_in[7];
  const float* gW1   = (const float*)d_in[8];
  const float* gb1   = (const float*)d_in[9];
  const float* gW2   = (const float*)d_in[10];
  const float* gb2   = (const float*)d_in[11];
  const float* temp  = (const float*)d_in[12];
  const float* epsf  = (const float*)d_in[13];
  const float* onw   = (const float*)d_in[14];
  const float* Wo    = (const float*)d_in[15];
  const float* fir_s = (const float*)d_in[16];
  const float* fir_l = (const float*)d_in[17];
  float* out = (float*)d_out;

  char* wsb = (char*)d_ws;
  size_t off = 0;
  auto alloc = [&](size_t bytes) -> void* {
    void* p = wsb + off;
    off = (off + bytes + 255) & ~(size_t)255;
    return p;
  };
  _Float16* xh   = (_Float16*)alloc((size_t)BL * HID * 2);
  _Float16* WqT  = (_Float16*)alloc((size_t)HID * 1024 * 2);
  _Float16* WkT  = (_Float16*)alloc((size_t)HID * 1024 * 2);
  _Float16* WvT  = (_Float16*)alloc((size_t)HID * 1024 * 2);
  _Float16* WoT  = (_Float16*)alloc((size_t)1024 * HID * 2);
  _Float16* gW1T = (_Float16*)alloc((size_t)GH * 5120 * 2);
  float* qp   = (float*)alloc((size_t)BL * 1024 * 4);
  float* kp   = (float*)alloc((size_t)BL * 1024 * 4);
  float* vp   = (float*)alloc((size_t)BL * 1024 * 4);
  float* qcv  = (float*)alloc((size_t)BL * 1024 * 4);
  float* kcv  = (float*)alloc((size_t)BL * 1024 * 4);
  float* vcv  = (float*)alloc((size_t)BL * 1024 * 4);
  float* beta = (float*)alloc((size_t)ROWS * 4);
  _Float16* q16  = (_Float16*)alloc((size_t)ROWS * DV * 2);
  _Float16* k16  = (_Float16*)alloc((size_t)ROWS * DV * 2);
  _Float16* u16g = (_Float16*)alloc((size_t)ROWS * DV * 2);
  _Float16* w16g = (_Float16*)alloc((size_t)ROWS * DV * 2);
  float* dlt  = (float*)alloc((size_t)ROWS * DV * 4);
  float* fso  = (float*)alloc((size_t)ROWS * DV * 4);
  float* flo  = (float*)alloc((size_t)ROWS * DV * 4);
  _Float16* fs16 = (_Float16*)alloc((size_t)ROWS * DV * 2);
  _Float16* fl16 = (_Float16*)alloc((size_t)ROWS * DV * 2);
  _Float16* dl16 = (_Float16*)alloc((size_t)ROWS * DV * 2);
  _Float16* v16b = (_Float16*)alloc((size_t)ROWS * DV * 2);
  float* stats = (float*)alloc((size_t)ROWS * 12 * 4);
  float* uvec  = (float*)alloc((size_t)12 * GH * 4);
  float* hmid  = (float*)alloc((size_t)ROWS * GH * 4);
  _Float16* omix = (_Float16*)alloc((size_t)BL * 1024 * 2);

  const int TPB = 256;
  auto cdiv = [](int a, int b) { return (a + b - 1) / b; };

  // 1) f16 conversions of activations + weight transposes (K-contiguous B operands)
  k_cvt_f16<<<cdiv(BL * HID, TPB), TPB, 0, stream>>>(x, xh, BL * HID);
  k_cvt_f16_T<<<cdiv(HID * 1024, TPB), TPB, 0, stream>>>(Wq, WqT, HID, 1024);
  k_cvt_f16_T<<<cdiv(HID * 1024, TPB), TPB, 0, stream>>>(Wk, WkT, HID, 1024);
  k_cvt_f16_T<<<cdiv(HID * 1024, TPB), TPB, 0, stream>>>(Wv, WvT, HID, 1024);
  k_cvt_f16_T<<<cdiv(1024 * HID, TPB), TPB, 0, stream>>>(Wo, WoT, 1024, HID);
  k_cvt_f16_T<<<cdiv(5120 * GH, TPB), TPB, 0, stream>>>(gW1, gW1T, 5120, GH);

  // 2) q/k/v projections: [2048,1024]@[1024,1024] via WMMA
  {
    int tiles = (BL / 16) * (1024 / 64);
    k_gemm<<<cdiv(tiles, 8), TPB, 0, stream>>>(xh, HID, 1, WqT, HID, 0, qp, 1024, BL, 1024, HID, 0);
    k_gemm<<<cdiv(tiles, 8), TPB, 0, stream>>>(xh, HID, 1, WkT, HID, 0, kp, 1024, BL, 1024, HID, 0);
    k_gemm<<<cdiv(tiles, 8), TPB, 0, stream>>>(xh, HID, 1, WvT, HID, 0, vp, 1024, BL, 1024, HID, 0);
  }
  // 3) beta + short convs + FIR branches
  k_beta<<<BL, 128, 0, stream>>>(x, Wb, beta);
  k_conv_silu<<<cdiv(BL * 1024, TPB), TPB, 0, stream>>>(qp, qc_w, qcv, 1024);
  k_conv_silu<<<cdiv(BL * 1024, TPB), TPB, 0, stream>>>(kp, kc_w, kcv, 1024);
  k_conv_silu<<<cdiv(BL * 1024, TPB), TPB, 0, stream>>>(vp, vc_w, vcv, 1024);
  k_fir<<<cdiv(ROWS * DV, TPB), TPB, 0, stream>>>(vcv, fir_s, fso, 3);
  k_fir<<<cdiv(ROWS * DV, TPB), TPB, 0, stream>>>(vcv, fir_l, flo, 31);

  // 4) delta rule: chunk precompute (256 blocks) + serial scan (8 blocks, S in 256KB LDS)
  k_delta_pre<<<Bb * NH * NCHUNK, TPB, 118784, stream>>>(qcv, kcv, vcv, beta, q16, k16, u16g, w16g);
  k_delta_scan<<<Bb * NH, TPB, 280576, stream>>>(q16, k16, u16g, w16g, dlt);

  // 5) gate MLP layer-1 via structure-exploiting decomposition
  k_cvt_f16<<<cdiv(ROWS * DV, TPB), TPB, 0, stream>>>(fso, fs16, ROWS * DV);
  k_cvt_f16<<<cdiv(ROWS * DV, TPB), TPB, 0, stream>>>(flo, fl16, ROWS * DV);
  k_cvt_f16<<<cdiv(ROWS * DV, TPB), TPB, 0, stream>>>(dlt, dl16, ROWS * DV);
  k_cvt_f16<<<cdiv(ROWS * DV, TPB), TPB, 0, stream>>>(vcv, v16b, ROWS * DV);
  k_colsum<<<cdiv(12 * GH, TPB), TPB, 0, stream>>>(gW1, uvec);
  k_stats<<<BL, TPB, 0, stream>>>(fso, flo, dlt, vcv, stats);
  k_hmid_init<<<cdiv(ROWS * GH, TPB), TPB, 0, stream>>>(stats, uvec, gb1, hmid);
  {
    int tiles = (ROWS / 16) * (GH / 64);
    // hidden block: rows of hmid (b,l,h) reuse xh row (b,l)  -> rowdiv = NH
    k_gemm<<<cdiv(tiles, 8), TPB, 0, stream>>>(xh, HID, NH, gW1T, 5120, 0, hmid, GH, ROWS, GH, HID, 1);
    k_gemm<<<cdiv(tiles, 8), TPB, 0, stream>>>(fs16, DV, 1, gW1T, 5120, 4096 + 0 * DV, hmid, GH, ROWS, GH, DV, 1);
    k_gemm<<<cdiv(tiles, 8), TPB, 0, stream>>>(fl16, DV, 1, gW1T, 5120, 4096 + 1 * DV, hmid, GH, ROWS, GH, DV, 1);
    k_gemm<<<cdiv(tiles, 8), TPB, 0, stream>>>(dl16, DV, 1, gW1T, 5120, 4096 + 2 * DV, hmid, GH, ROWS, GH, DV, 1);
    k_gemm<<<cdiv(tiles, 8), TPB, 0, stream>>>(v16b, DV, 1, gW1T, 5120, 4096 + 3 * DV, hmid, GH, ROWS, GH, DV, 1);
  }
  // 6) gate head + mixing + RMS norm -> omix (f16)
  k_gate_mix<<<BL, 128, 0, stream>>>(hmid, gW2, gb2, temp, epsf, onw, fso, flo, dlt, vcv, omix);
  // 7) output projection -> d_out (f32)
  {
    int tiles = (BL / 16) * (HID / 64);
    k_gemm<<<cdiv(tiles, 8), TPB, 0, stream>>>(omix, NH * DV, 1, WoT, 1024, 0, out, HID, BL, HID, 1024, 0);
  }
}
